// Attention_65189013619346
// MI455X (gfx1250) — matrix operand order repaired and checked by reading, not compiled, not run on hardware
//
#include <hip/hip_runtime.h>
#include <hip/hip_bf16.h>
#include <stdint.h>

typedef __bf16 bf16_t;
typedef __attribute__((ext_vector_type(16))) __bf16 v16bf;
typedef __attribute__((ext_vector_type(8)))  __bf16 v8bf;
typedef __attribute__((ext_vector_type(8)))  float  v8f;

#define HEADS 8
#define DKEY  64
#define DVAL  192
#define DIMM  1536
#define BATCH 2
#define SEQ   2048
#define NRP   4095   /* 2*SEQ-1 */

// ---------------------------------------------------------------------------
// WMMA helpers (CDNA5 wave32, V_WMMA_F32_16X16X32_BF16)
// ---------------------------------------------------------------------------
__device__ __forceinline__ v8f wmma_bf16(v16bf a, v16bf b, v8f c) {
    return __builtin_amdgcn_wmma_f32_16x16x32_bf16(
        /*neg_a=*/false, a, /*neg_b=*/false, b,
        /*c_mod=*/(short)0, c, /*reuse_a=*/false, /*reuse_b=*/false);
}

__device__ __forceinline__ v16bf cat8(v8bf lo, v8bf hi) {
    return __builtin_shufflevector(lo, hi, 0, 1, 2, 3, 4, 5, 6, 7,
                                           8, 9, 10, 11, 12, 13, 14, 15);
}

// A fragment: 16x32 bf16 tile, row-major source, leading dim ld (elements).
// Layout: lanes 0-15 row M=lane hold K=0..15; lanes 16-31 hold
// K=16..31. Both halves are contiguous 16-byte runs -> two b128 loads.
__device__ __forceinline__ v16bf load_frag_a(const bf16_t* p, int ld) {
    const int lane = threadIdx.x & 31;
    const int m  = lane & 15;
    const int kh = (lane >> 4) << 4;                 // 0 or 16
    const bf16_t* base = p + (size_t)m * ld + kh;
    const v8bf lo = *(const v8bf*)(base);            // K = kh .. kh+7
    const v8bf hi = *(const v8bf*)(base + 8);        // K = kh+8 .. kh+15
    return cat8(lo, hi);
}

// B fragment where B[k][n] = p[n*ld + k] (K^T / relk^T / V^T style storage).
// Each lane's 16 K-values are contiguous -> single 32-byte load.
__device__ __forceinline__ v16bf load_frag_bT(const bf16_t* p, int ld) {
    const int lane = threadIdx.x & 31;
    const int n  = lane & 15;
    const int ko = (lane >> 4) << 4;                 // 0 or 16
    return *(const v16bf*)(p + (size_t)n * ld + ko);
}

// XOR-butterfly reductions within 16-lane row groups via ds_swizzle
// (group-of-32 mode: and_mask=31, xor_mask in bits [14:10]).
__device__ __forceinline__ float bcast_max16(float x) {
    float y;
    y = __int_as_float(__builtin_amdgcn_ds_swizzle(__float_as_int(x), 0x041F)); x = fmaxf(x, y);
    y = __int_as_float(__builtin_amdgcn_ds_swizzle(__float_as_int(x), 0x081F)); x = fmaxf(x, y);
    y = __int_as_float(__builtin_amdgcn_ds_swizzle(__float_as_int(x), 0x101F)); x = fmaxf(x, y);
    y = __int_as_float(__builtin_amdgcn_ds_swizzle(__float_as_int(x), 0x201F)); x = fmaxf(x, y);
    return x;
}
__device__ __forceinline__ float bcast_sum16(float x) {
    x += __int_as_float(__builtin_amdgcn_ds_swizzle(__float_as_int(x), 0x041F));
    x += __int_as_float(__builtin_amdgcn_ds_swizzle(__float_as_int(x), 0x081F));
    x += __int_as_float(__builtin_amdgcn_ds_swizzle(__float_as_int(x), 0x101F));
    x += __int_as_float(__builtin_amdgcn_ds_swizzle(__float_as_int(x), 0x201F));
    return x;
}

// ---------------------------------------------------------------------------
// Elementwise fp32 -> bf16 cast (row-major preserving)
// ---------------------------------------------------------------------------
__global__ void f32_to_bf16_kernel(const float* __restrict__ src,
                                   bf16_t* __restrict__ dst, int count) {
    int i = blockIdx.x * blockDim.x + threadIdx.x;
    if (i < count) dst[i] = (bf16_t)src[i];
}

// fp32 [R][C] -> bf16 transposed [C][R]; output-coalesced.
__global__ void f32_to_bf16_T_kernel(const float* __restrict__ src,
                                     bf16_t* __restrict__ dst, int R, int C) {
    int i = blockIdx.x * blockDim.x + threadIdx.x;
    if (i >= R * C) return;
    const int n = i / R;       // output row (= source column)
    const int k = i % R;       // output col (= source row)
    dst[i] = (bf16_t)src[(size_t)k * C + n];
}

// ---------------------------------------------------------------------------
// LDS-free WMMA GEMM: C[M,N] f32 (+bias) = A[M,K]bf16 @ Bt[N,K]bf16^T
// A row-major, Bt transposed (N-major). Block: 128 thr (4 waves);
// block tile 128x64, wave tile 32x64 (B frags reused x2, A frags x4).
// K-loop unrolled x2 with explicit ping-pong fragment sets: every fragment
// register is written by exactly one load and read by its wmmas -> no buffer
// rotation copies, loads for one panel overlap the other panel's wmmas.
// Requires M%128==0, N%64==0, K%64==0.
// ---------------------------------------------------------------------------
__global__ void __launch_bounds__(128)
gemm_bf16_kernel(const bf16_t* __restrict__ A, const bf16_t* __restrict__ Bt,
                 float* __restrict__ C, const float* __restrict__ bias,
                 int M, int N, int K) {
    const int bm   = blockIdx.y * 128;
    const int bn   = blockIdx.x * 64;
    const int wave = threadIdx.x >> 5;
    const int lane = threadIdx.x & 31;

    const bf16_t* aRow0 = A + (size_t)(bm + wave * 32) * K;
    const bf16_t* aRow1 = aRow0 + (size_t)16 * K;
    const bf16_t* bt0 = Bt + (size_t)(bn +  0) * K;
    const bf16_t* bt1 = Bt + (size_t)(bn + 16) * K;
    const bf16_t* bt2 = Bt + (size_t)(bn + 32) * K;
    const bf16_t* bt3 = Bt + (size_t)(bn + 48) * K;

    v8f acc0[4] = {};
    v8f acc1[4] = {};

    // prologue: even panel (k = 0)
    v16bf a0A = load_frag_a(aRow0, K);
    v16bf a1A = load_frag_a(aRow1, K);
    v16bf bA0 = load_frag_bT(bt0, K), bA1 = load_frag_bT(bt1, K);
    v16bf bA2 = load_frag_bT(bt2, K), bA3 = load_frag_bT(bt3, K);

    for (int k0 = 0; k0 < K; k0 += 64) {
        // issue odd-panel loads (k0+32); overlap with even-panel wmmas
        const int kB = k0 + 32;
        __builtin_prefetch(aRow0 + (size_t)(lane & 15) * K + k0 + 64, 0, 1);
        const v16bf a0B = load_frag_a(aRow0 + kB, K);
        const v16bf a1B = load_frag_a(aRow1 + kB, K);
        const v16bf bB0 = load_frag_bT(bt0 + kB, K);
        const v16bf bB1 = load_frag_bT(bt1 + kB, K);
        const v16bf bB2 = load_frag_bT(bt2 + kB, K);
        const v16bf bB3 = load_frag_bT(bt3 + kB, K);

        acc0[0] = wmma_bf16(a0A, bA0, acc0[0]);
        acc1[0] = wmma_bf16(a1A, bA0, acc1[0]);
        acc0[1] = wmma_bf16(a0A, bA1, acc0[1]);
        acc1[1] = wmma_bf16(a1A, bA1, acc1[1]);
        acc0[2] = wmma_bf16(a0A, bA2, acc0[2]);
        acc1[2] = wmma_bf16(a1A, bA2, acc1[2]);
        acc0[3] = wmma_bf16(a0A, bA3, acc0[3]);
        acc1[3] = wmma_bf16(a1A, bA3, acc1[3]);

        // issue next even-panel loads (k0+64); overlap with odd-panel wmmas
        const int kA = k0 + 64;
        if (kA < K) {
            a0A = load_frag_a(aRow0 + kA, K);
            a1A = load_frag_a(aRow1 + kA, K);
            bA0 = load_frag_bT(bt0 + kA, K);
            bA1 = load_frag_bT(bt1 + kA, K);
            bA2 = load_frag_bT(bt2 + kA, K);
            bA3 = load_frag_bT(bt3 + kA, K);
        }

        acc0[0] = wmma_bf16(a0B, bB0, acc0[0]);
        acc1[0] = wmma_bf16(a1B, bB0, acc1[0]);
        acc0[1] = wmma_bf16(a0B, bB1, acc0[1]);
        acc1[1] = wmma_bf16(a1B, bB1, acc1[1]);
        acc0[2] = wmma_bf16(a0B, bB2, acc0[2]);
        acc1[2] = wmma_bf16(a1B, bB2, acc1[2]);
        acc0[3] = wmma_bf16(a0B, bB3, acc0[3]);
        acc1[3] = wmma_bf16(a1B, bB3, acc1[3]);
    }

    const int cn = lane & 15;
    const int rh = (lane >> 4) << 3;
#pragma unroll
    for (int t = 0; t < 4; ++t)
#pragma unroll
        for (int v = 0; v < 8; ++v) {
            const int c = bn + t * 16 + cn;
            const float bv = bias ? bias[c] : 0.f;
            const int r0 = bm + wave * 32 + v + rh;
            C[(size_t)r0 * N + c] = acc0[t][v] + bv;
            const int r1 = r0 + 16;
            C[(size_t)r1 * N + c] = acc1[t][v] + bv;
        }
}

// ---------------------------------------------------------------------------
// q/k finalize: fold scale + rel_content_bias / rel_pos_bias, transpose to
// [b,h,n,dk] bf16.
// ---------------------------------------------------------------------------
__global__ void qk_finalize_kernel(const float* __restrict__ qf,
                                   const float* __restrict__ kf,
                                   const float* __restrict__ rcb,
                                   const float* __restrict__ rpb,
                                   bf16_t* __restrict__ qc,
                                   bf16_t* __restrict__ qr,
                                   bf16_t* __restrict__ kb) {
    const int total = BATCH * SEQ * HEADS * DKEY;
    int i = blockIdx.x * blockDim.x + threadIdx.x;
    if (i >= total) return;
    const int c   = i % (HEADS * DKEY);
    const int seq = (i / (HEADS * DKEY)) % SEQ;
    const int b   = i / (SEQ * HEADS * DKEY);
    const int h   = c / DKEY, d = c % DKEY;
    const float scale = 0.125f;  // 64^-0.5
    const size_t o = (((size_t)(b * HEADS + h)) * SEQ + seq) * DKEY + d;
    const float q = qf[i] * scale;
    qc[o] = (bf16_t)(q + rcb[c]);
    qr[o] = (bf16_t)(q + rpb[c]);
    kb[o] = (bf16_t)kf[i];
}

// v: [b,seq,h*dv] f32 -> V^T bf16 [b,h,dv,seq]  (output-coalesced over seq)
__global__ void v_finalize_T_kernel(const float* __restrict__ vf,
                                    bf16_t* __restrict__ vt) {
    const int total = BATCH * SEQ * HEADS * DVAL;
    int i = blockIdx.x * blockDim.x + threadIdx.x;
    if (i >= total) return;
    const int seq = i % SEQ;
    const int d   = (i / SEQ) % DVAL;
    const int bh  = i / (SEQ * DVAL);
    const int b   = bh / HEADS, h = bh % HEADS;
    vt[i] = (bf16_t)vf[((size_t)b * SEQ + seq) * (HEADS * DVAL) + h * DVAL + d];
}

// ---------------------------------------------------------------------------
// relk[h][p][d] = (positional_embed(p) @ Wrel) in bf16, positions computed
// in closed form (central-mask basis + signed copy).
// ---------------------------------------------------------------------------
__global__ void relk_kernel(const float* __restrict__ Wrel,
                            bf16_t* __restrict__ relk) {
    const int total = NRP * HEADS * DKEY;
    int i = blockIdx.x * blockDim.x + threadIdx.x;
    if (i >= total) return;
    const int c = i % (HEADS * DKEY);
    const int p = i / (HEADS * DKEY);
    const int h = c / DKEY, d = c % DKEY;
    const float dist = (float)(p - (SEQ - 1));
    const float ad = fabsf(dist);
    const float sg = (dist > 0.f) ? 1.f : ((dist < 0.f) ? -1.f : 0.f);
    const float rate = __expf(__logf((float)(SEQ + 1)) / 16.f);
    float acc = 0.f;
    float cw = 1.f;
#pragma unroll
    for (int j = 0; j < 16; ++j) {
        cw *= rate;                                  // rate^(j+1)
        const float e = ((cw - 1.f) > ad) ? 1.f : 0.f;
        acc += e * Wrel[j * (HEADS * DKEY) + c];
        acc += sg * e * Wrel[(j + 16) * (HEADS * DKEY) + c];
    }
    relk[((size_t)h * NRP + p) * DKEY + d] = (bf16_t)acc;
}

// ---------------------------------------------------------------------------
// Fused flash attention with relative-shift gather.
// grid = (SEQ/64, BATCH*HEADS), block = 128 (4 waves); wave owns 16 rows.
// shifted_rel[i,j] = rel_dot[i, j-i+n-1]  (Transformer-XL shift identity).
// ---------------------------------------------------------------------------
__global__ void __launch_bounds__(128)
flash_attn_kernel(const bf16_t* __restrict__ qc, const bf16_t* __restrict__ qr,
                  const bf16_t* __restrict__ kb, const bf16_t* __restrict__ vt,
                  const bf16_t* __restrict__ relk, bf16_t* __restrict__ obf) {
    __shared__ float  sR[4 * 16 * 128];   // per-wave Rfull slab (32 KB)
    __shared__ bf16_t sP[4 * 16 * 64];    // per-wave P staging   (8 KB)

    const int i0   = blockIdx.x * 64;
    const int bh   = blockIdx.y;
    const int b    = bh / HEADS, h = bh % HEADS;
    const int wave = threadIdx.x >> 5;
    const int lane = threadIdx.x & 31;
    const int cn   = lane & 15;
    const int rh   = (lane >> 4) << 3;

    const bf16_t* qcw   = qc + ((size_t)bh * SEQ + i0 + wave * 16) * DKEY;
    const bf16_t* qrw   = qr + ((size_t)bh * SEQ + i0 + wave * 16) * DKEY;
    const bf16_t* kbase = kb + (size_t)bh * SEQ * DKEY;
    const bf16_t* vbase = vt + (size_t)bh * DVAL * SEQ;   // V^T: [d][seq]
    const bf16_t* rbase = relk + (size_t)h * NRP * DKEY;
    float*  sRw = sR + wave * 16 * 128;
    bf16_t* sPw = sP + wave * 16 * 64;

    // Persistent query fragments (K = 64 -> two 16x32 frags each)
    v16bf aQc[2], aQr[2];
    aQc[0] = load_frag_a(qcw, DKEY);       aQc[1] = load_frag_a(qcw + 32, DKEY);
    aQr[0] = load_frag_a(qrw, DKEY);       aQr[1] = load_frag_a(qrw + 32, DKEY);

    float m_state[8], l_state[8];
#pragma unroll
    for (int v = 0; v < 8; ++v) { m_state[v] = -3.0e38f; l_state[v] = 0.f; }
    v8f Oacc[12] = {};

    for (int j0 = 0; j0 < SEQ; j0 += 64) {
        // ---- relative logits slice: Rfull[row, t] = q_r . relk[base0 + t]
        const int base0 = j0 - i0 + (SEQ - 1) - 63;       // in [0, 3968]
        const bf16_t* rp = rbase + (size_t)base0 * DKEY;
        {
            v8f Racc[8] = {};
#pragma unroll
            for (int ks = 0; ks < 2; ++ks)
#pragma unroll
                for (int rt = 0; rt < 8; ++rt)
                    Racc[rt] = wmma_bf16(aQr[ks],
                        load_frag_bT(rp + rt * 16 * DKEY + ks * 32, DKEY),
                        Racc[rt]);
#pragma unroll
            for (int rt = 0; rt < 8; ++rt)
#pragma unroll
                for (int v = 0; v < 8; ++v)
                    sRw[(v + rh) * 128 + rt * 16 + cn] = Racc[rt][v];
        }

        // ---- content logits tile S = q_c @ K^T
        v8f Sacc[4] = {};
        const bf16_t* kp = kbase + (size_t)j0 * DKEY;
#pragma unroll
        for (int ks = 0; ks < 2; ++ks)
#pragma unroll
            for (int jt = 0; jt < 4; ++jt)
                Sacc[jt] = wmma_bf16(aQc[ks],
                    load_frag_bT(kp + jt * 16 * DKEY + ks * 32, DKEY),
                    Sacc[jt]);

        // ---- shift-gather + combine
        float L[4][8];
#pragma unroll
        for (int jt = 0; jt < 4; ++jt)
#pragma unroll
            for (int v = 0; v < 8; ++v) {
                const int ii = wave * 16 + v + rh;          // tile-local row
                const int jj = jt * 16 + cn;
                const int t  = jj - ii + 63;                // in [0, 126]
                L[jt][v] = Sacc[jt][v] + sRw[(v + rh) * 128 + t];
            }

        // ---- online softmax (per-row state; 16-lane XOR butterflies)
#pragma unroll
        for (int v = 0; v < 8; ++v) {
            float mx = fmaxf(fmaxf(L[0][v], L[1][v]), fmaxf(L[2][v], L[3][v]));
            mx = bcast_max16(mx);
            const float mnew  = fmaxf(m_state[v], mx);
            const float alpha = __expf(m_state[v] - mnew);
            float psum = 0.f;
#pragma unroll
            for (int jt = 0; jt < 4; ++jt) {
                const float pe = __expf(L[jt][v] - mnew);
                sPw[(v + rh) * 64 + jt * 16 + cn] = (bf16_t)pe;
                psum += pe;
            }
            psum = bcast_sum16(psum);
            l_state[v] = l_state[v] * alpha + psum;
            m_state[v] = mnew;
#pragma unroll
            for (int vt2 = 0; vt2 < 12; ++vt2) Oacc[vt2][v] *= alpha;
        }

        // ---- O += P @ V   (P restaged via per-wave LDS; V^T gives
        //      contiguous-in-seq b128 fragment loads)
#pragma unroll
        for (int ks = 0; ks < 2; ++ks) {
            const v16bf aP = load_frag_a(sPw + ks * 32, 64);
#pragma unroll
            for (int vt2 = 0; vt2 < 12; ++vt2)
                Oacc[vt2] = wmma_bf16(aP,
                    load_frag_bT(vbase + (size_t)(vt2 * 16) * SEQ + j0 + ks * 32,
                                 SEQ),
                    Oacc[vt2]);
        }
    }

    // ---- normalize and emit bf16 in [b, seq, h*dv] for the output GEMM
#pragma unroll
    for (int vt2 = 0; vt2 < 12; ++vt2)
#pragma unroll
        for (int v = 0; v < 8; ++v) {
            const int row = i0 + wave * 16 + v + rh;
            const int col = h * DVAL + vt2 * 16 + cn;
            const float val = Oacc[vt2][v] / l_state[v];
            obf[((size_t)b * SEQ + row) * (HEADS * DVAL) + col] = (bf16_t)val;
        }
}

// ---------------------------------------------------------------------------
// Host orchestration
// ---------------------------------------------------------------------------
extern "C" void kernel_launch(void* const* d_in, const int* in_sizes, int n_in,
                              void* d_out, int out_size, void* d_ws, size_t ws_size,
                              hipStream_t stream) {
    (void)in_sizes; (void)n_in; (void)out_size; (void)ws_size;
    const float* x    = (const float*)d_in[0];
    const float* Wq   = (const float*)d_in[1];
    const float* Wk   = (const float*)d_in[2];
    const float* Wv   = (const float*)d_in[3];
    const float* Wrel = (const float*)d_in[4];
    const float* rcb  = (const float*)d_in[5];
    const float* rpb  = (const float*)d_in[6];
    const float* Wout = (const float*)d_in[7];
    const float* bout = (const float*)d_in[8];
    float* out = (float*)d_out;

    char* ws = (char*)d_ws;
    size_t off = 0;
    auto alloc = [&](size_t bytes) -> void* {
        off = (off + 255) & ~(size_t)255;
        void* p = ws + off;
        off += bytes;
        return p;
    };

    const size_t MR = (size_t)BATCH * SEQ;  // 4096
    bf16_t* x_bf    = (bf16_t*)alloc(MR * DIMM * 2);
    bf16_t* WqT     = (bf16_t*)alloc((size_t)DIMM * HEADS * DKEY * 2);  // [N][K]
    bf16_t* WkT     = (bf16_t*)alloc((size_t)DIMM * HEADS * DKEY * 2);
    bf16_t* WvT     = (bf16_t*)alloc((size_t)DIMM * HEADS * DVAL * 2);
    bf16_t* WoutT   = (bf16_t*)alloc((size_t)HEADS * DVAL * DIMM * 2);
    float*  q_f     = (float*)alloc(MR * HEADS * DKEY * 4);
    float*  k_f     = (float*)alloc(MR * HEADS * DKEY * 4);
    float*  v_f     = (float*)alloc(MR * HEADS * DVAL * 4);
    bf16_t* q_c     = (bf16_t*)alloc(MR * HEADS * DKEY * 2);
    bf16_t* q_r     = (bf16_t*)alloc(MR * HEADS * DKEY * 2);
    bf16_t* k_b     = (bf16_t*)alloc(MR * HEADS * DKEY * 2);
    bf16_t* v_t     = (bf16_t*)alloc(MR * HEADS * DVAL * 2);            // V^T
    bf16_t* relk    = (bf16_t*)alloc(((size_t)HEADS * NRP + 128) * DKEY * 2);
    bf16_t* o_bf    = (bf16_t*)alloc(MR * HEADS * DVAL * 2);

    // activations: straight cast; weights: cast + transpose to [N][K]
    {
        const int nx = (int)(MR * DIMM);
        f32_to_bf16_kernel<<<(nx + 255) / 256, 256, 0, stream>>>(x, x_bf, nx);
        const int nqk = DIMM * HEADS * DKEY;
        f32_to_bf16_T_kernel<<<(nqk + 255) / 256, 256, 0, stream>>>(Wq, WqT, DIMM, HEADS * DKEY);
        f32_to_bf16_T_kernel<<<(nqk + 255) / 256, 256, 0, stream>>>(Wk, WkT, DIMM, HEADS * DKEY);
        const int nv = DIMM * HEADS * DVAL;
        f32_to_bf16_T_kernel<<<(nv + 255) / 256, 256, 0, stream>>>(Wv, WvT, DIMM, HEADS * DVAL);
        const int no = HEADS * DVAL * DIMM;
        f32_to_bf16_T_kernel<<<(no + 255) / 256, 256, 0, stream>>>(Wout, WoutT, HEADS * DVAL, DIMM);
    }

    const dim3 blk(128);
    gemm_bf16_kernel<<<dim3((HEADS * DKEY) / 64, (unsigned)(MR / 128)), blk, 0, stream>>>(
        x_bf, WqT, q_f, nullptr, (int)MR, HEADS * DKEY, DIMM);
    gemm_bf16_kernel<<<dim3((HEADS * DKEY) / 64, (unsigned)(MR / 128)), blk, 0, stream>>>(
        x_bf, WkT, k_f, nullptr, (int)MR, HEADS * DKEY, DIMM);
    gemm_bf16_kernel<<<dim3((HEADS * DVAL) / 64, (unsigned)(MR / 128)), blk, 0, stream>>>(
        x_bf, WvT, v_f, nullptr, (int)MR, HEADS * DVAL, DIMM);

    {
        const int nqk = (int)(MR * HEADS * DKEY);
        qk_finalize_kernel<<<(nqk + 255) / 256, 256, 0, stream>>>(
            q_f, k_f, rcb, rpb, q_c, q_r, k_b);
        const int nv = (int)(MR * HEADS * DVAL);
        v_finalize_T_kernel<<<(nv + 255) / 256, 256, 0, stream>>>(v_f, v_t);
        const int nr = NRP * HEADS * DKEY;
        relk_kernel<<<(nr + 255) / 256, 256, 0, stream>>>(Wrel, relk);
    }

    flash_attn_kernel<<<dim3(SEQ / 64, BATCH * HEADS), blk, 0, stream>>>(
        q_c, q_r, k_b, v_t, relk, o_bf);

    gemm_bf16_kernel<<<dim3(DIMM / 64, (unsigned)(MR / 128)), blk, 0, stream>>>(
        o_bf, WoutT, out, bout, (int)MR, DIMM, HEADS * DVAL);
}
